// TransformerBlock_6700148982342
// MI455X (gfx1250) — compile-verified, matrix operand order checked by reading, not run on hardware
//
#include <hip/hip_runtime.h>

// ---------------------------------------------------------------------------
// Transformer block for MI455X (gfx1250): bf16 WMMA GEMMs + flash attention.
// fp32 accumulate everywhere; bf16 only on matmul operands.
// ---------------------------------------------------------------------------

#define N_TOK 8192   // B*P
#define BATCH 4
#define SEQ   2048
#define CDIM  1024
#define HID   4096
#define NHEAD 16
#define HDIM  64

typedef __bf16 v16bf __attribute__((ext_vector_type(16)));
typedef float  v8f   __attribute__((ext_vector_type(8)));
typedef unsigned int u32;
typedef __attribute__((ext_vector_type(4))) unsigned int u32x4;
typedef __attribute__((ext_vector_type(4))) int i32x4;

union AF { v16bf v; u32x4 q[2]; };
union CF { v8f   v; float f[8]; };
union US8 { u32x4 q; unsigned short s[8]; };

#ifdef __HIP_DEVICE_COMPILE__
#define DEV_ASM(s) asm volatile(s ::: "memory")
#else
#define DEV_ASM(s)
#endif

// --- async global->LDS (CDNA5 GLOBAL_LOAD_ASYNC_TO_LDS_B128), guarded ------
#if defined(__HIP_DEVICE_COMPILE__) && defined(__has_builtin)
#if __has_builtin(__builtin_amdgcn_global_load_async_to_lds_b128)
#define USE_ASYNC_LDS 1
#endif
#endif

__device__ __forceinline__ void cp16(unsigned short* lds_dst,
                                     const unsigned short* gsrc) {
#ifdef USE_ASYNC_LDS
  // Param types per hipcc diagnostic: (v4i AS1*, v4i AS3*, imm, imm).
  __builtin_amdgcn_global_load_async_to_lds_b128(
      (__attribute__((address_space(1))) i32x4*)(unsigned long long)gsrc,
      (__attribute__((address_space(3))) i32x4*)(unsigned long long)lds_dst,
      0, 0);
#else
  *(u32x4*)lds_dst = *(const u32x4*)gsrc;
#endif
}

__device__ __forceinline__ void async_wait() {
#ifdef USE_ASYNC_LDS
#if __has_builtin(__builtin_amdgcn_s_wait_asynccnt)
  __builtin_amdgcn_s_wait_asynccnt(0);
#else
  DEV_ASM("s_wait_asynccnt 0x0");
#endif
#endif
}

// f32 -> bf16 (round-to-nearest-even), bit-level so no scalar __bf16 math.
__device__ __forceinline__ unsigned short f2bf(float f) {
  union { float f; u32 u; } v; v.f = f;
  u32 r = v.u + 0x7FFFu + ((v.u >> 16) & 1u);
  if ((v.u & 0x7F800000u) == 0x7F800000u) r = v.u;  // inf/nan passthrough
  return (unsigned short)(r >> 16);
}

// ---------------------------------------------------------------------------
// f32 -> bf16 elementwise convert (weights)
// ---------------------------------------------------------------------------
__global__ __launch_bounds__(256) void k_cvt_bf16(const float* __restrict__ in,
                                                  unsigned short* __restrict__ out,
                                                  int n) {
  int i = blockIdx.x * 256 + threadIdx.x;
  if (i < n) out[i] = f2bf(in[i]);
}

// ---------------------------------------------------------------------------
// LayerNorm over C=1024, one block per token row, bf16 output.
// ---------------------------------------------------------------------------
__global__ __launch_bounds__(256) void k_layernorm(const float* __restrict__ x,
                                                   const float* __restrict__ gw,
                                                   const float* __restrict__ bw,
                                                   unsigned short* __restrict__ out) {
  __shared__ float red[8];
  const int row = blockIdx.x;
  const int lane = threadIdx.x & 31, wave = threadIdx.x >> 5;
  const float* xr = x + (size_t)row * CDIM;
  float v[4]; float s = 0.f;
#pragma unroll
  for (int i = 0; i < 4; i++) { v[i] = xr[threadIdx.x + i * 256]; s += v[i]; }
#pragma unroll
  for (int m = 16; m; m >>= 1) s += __shfl_xor(s, m, 32);
  if (lane == 0) red[wave] = s;
  __syncthreads();
  s = red[0] + red[1] + red[2] + red[3] + red[4] + red[5] + red[6] + red[7];
  const float mean = s * (1.f / CDIM);
  float vs = 0.f;
#pragma unroll
  for (int i = 0; i < 4; i++) { float d = v[i] - mean; vs += d * d; }
#pragma unroll
  for (int m = 16; m; m >>= 1) vs += __shfl_xor(vs, m, 32);
  __syncthreads();
  if (lane == 0) red[wave] = vs;
  __syncthreads();
  vs = red[0] + red[1] + red[2] + red[3] + red[4] + red[5] + red[6] + red[7];
  const float inv = rsqrtf(vs * (1.f / CDIM) + 1e-6f);
#pragma unroll
  for (int i = 0; i < 4; i++) {
    int c = threadIdx.x + i * 256;
    out[(size_t)row * CDIM + c] = f2bf((v[i] - mean) * inv * gw[c] + bw[c]);
  }
}

// ---------------------------------------------------------------------------
// bf16 WMMA GEMM:  out = epilogue( A[N x K] @ W[K x M] )
// Block tile 128x128, 8 waves (2x4), each wave 64x32 (4x2 wmma tiles), BK=64.
// Epilogues:
//   0: +bias -> bf16, scattered to head-major [b,h,t,d] (QKV)
//   1: +bias +residual -> f32
//   2: +bias, bn, exact-erf GELU -> bf16
//   3: +bias, bn, +residual -> f32
// ---------------------------------------------------------------------------
#define BM 128
#define BN 128
#define BK 64
#define BKP (BK + 8)

template <int EPI>
__global__ __launch_bounds__(256) void k_gemm_bf16(
    const unsigned short* __restrict__ A, const unsigned short* __restrict__ W,
    const float* __restrict__ bias,
    const float* __restrict__ bn_g, const float* __restrict__ bn_b,
    const float* __restrict__ bn_m, const float* __restrict__ bn_v,
    const float* __restrict__ resid, void* __restrict__ outp,
    int K, int M) {
  __shared__ __align__(16) unsigned short As[BM][BKP];
  __shared__ __align__(16) unsigned short Bs[BN][BKP];

  const int tid = threadIdx.x;
  const int wave = tid >> 5, lane = tid & 31;
  const int r = lane & 15, half = lane >> 4;
  const int wm = wave & 1, wn = wave >> 1;  // 2 x 4 wave grid
  const int row0 = blockIdx.y * BM;
  const int col0 = blockIdx.x * BN;

  CF acc[4][2];
#pragma unroll
  for (int mi = 0; mi < 4; mi++)
#pragma unroll
    for (int ni = 0; ni < 2; ni++) acc[mi][ni].v = (v8f){0, 0, 0, 0, 0, 0, 0, 0};

  for (int k0 = 0; k0 < K; k0 += BK) {
    // A tile: 128x64 bf16, row-major, async b128 copies (4 per thread).
#pragma unroll
    for (int i = 0; i < 4; i++) {
      int idx = tid + i * 256;
      int arow = idx >> 3, acw = idx & 7;
      cp16(&As[arow][acw * 8], A + (size_t)(row0 + arow) * K + k0 + acw * 8);
    }
    // W tile: 64 x 128, stored transposed in LDS -> Bs[n][k].
#pragma unroll
    for (int i = 0; i < 4; i++) {
      int idx = tid + i * 256;
      int kk = idx >> 4, ncw = idx & 15;
      US8 d; d.q = *(const u32x4*)(W + (size_t)(k0 + kk) * M + col0 + ncw * 8);
#pragma unroll
      for (int e = 0; e < 8; e++) Bs[ncw * 8 + e][kk] = d.s[e];
    }
    if (k0 + BK < K) {  // hint next tiles into cache (global_prefetch_b8)
      __builtin_prefetch(A + (size_t)(row0 + (tid >> 1)) * K + k0 + BK, 0, 1);
      __builtin_prefetch(W + (size_t)(k0 + BK + (tid >> 2)) * M + col0, 0, 1);
    }
    async_wait();
    __syncthreads();

#pragma unroll
    for (int kc = 0; kc < 2; kc++) {
      AF a[4], b[2];
#pragma unroll
      for (int mi = 0; mi < 4; mi++) {
        const unsigned short* p = &As[wm * 64 + mi * 16 + r][kc * 32 + half * 8];
        a[mi].q[0] = *(const u32x4*)p;
        a[mi].q[1] = *(const u32x4*)(p + 16);
      }
#pragma unroll
      for (int ni = 0; ni < 2; ni++) {
        const unsigned short* p = &Bs[wn * 32 + ni * 16 + r][kc * 32 + half * 8];
        b[ni].q[0] = *(const u32x4*)p;
        b[ni].q[1] = *(const u32x4*)(p + 16);
      }
#pragma unroll
      for (int mi = 0; mi < 4; mi++)
#pragma unroll
        for (int ni = 0; ni < 2; ni++)
          acc[mi][ni].v = __builtin_amdgcn_wmma_f32_16x16x32_bf16(
              false, a[mi].v, false, b[ni].v, (short)0, acc[mi][ni].v, false, false);
    }
    __syncthreads();
  }

  // Epilogue. C/D layout: element i -> row (i + half*8), col r.
#pragma unroll
  for (int ni = 0; ni < 2; ni++) {
    const int col = col0 + wn * 32 + ni * 16 + r;
    const float bia = bias ? bias[col] : 0.f;
    float sg = 0.f, sb = 0.f, sm = 0.f, sv = 0.f;
    if (EPI == 2 || EPI == 3) {
      sg = bn_g[col]; sb = bn_b[col]; sm = bn_m[col];
      sv = rsqrtf(bn_v[col] + 1e-6f);
    }
#pragma unroll
    for (int mi = 0; mi < 4; mi++) {
#pragma unroll
      for (int i = 0; i < 8; i++) {
        const int row = row0 + wm * 64 + mi * 16 + i + half * 8;
        float y = acc[mi][ni].f[i] + bia;
        if (EPI == 0) {
          // ref reshape (B,P,HD,HEADS): channel c -> h = c%16, d = c/16.
          const int h = col & 15, d = col >> 4;
          const int bb = row >> 11, t = row & 2047;
          ((unsigned short*)outp)[(((size_t)(bb * NHEAD + h)) * SEQ + t) * HDIM + d] = f2bf(y);
        } else if (EPI == 1) {
          ((float*)outp)[(size_t)row * M + col] = y + resid[(size_t)row * M + col];
        } else if (EPI == 2) {
          y = (y - sm) * sv * sg + sb;
          y = 0.5f * y * (1.f + erff(y * 0.70710678118654752f));  // exact gelu
          ((unsigned short*)outp)[(size_t)row * M + col] = f2bf(y);
        } else {
          y = (y - sm) * sv * sg + sb;
          ((float*)outp)[(size_t)row * M + col] = y + resid[(size_t)row * M + col];
        }
      }
    }
  }
}

// ---------------------------------------------------------------------------
// Flash attention, one block per (b, h, 128-query tile); 8 waves x 16 queries.
// Q/K/V in head-major [b,h,t,d] bf16; output in concat layout [b,t, h*64+d].
// ---------------------------------------------------------------------------
__global__ __launch_bounds__(256) void k_flash_attn(
    const unsigned short* __restrict__ qh, const unsigned short* __restrict__ kh,
    const unsigned short* __restrict__ vh, unsigned short* __restrict__ outp) {
  __shared__ __align__(16) unsigned short Kt[128][BK + 8];        // [key][dim]
  __shared__ __align__(16) unsigned short Vt[HDIM][128 + 8];      // [dim][key]
  __shared__ __align__(16) unsigned short Ps[8][16][128 + 8];     // per-wave P

  const int tid = threadIdx.x;
  const int wave = tid >> 5, lane = tid & 31;
  const int r = lane & 15, half = lane >> 4;
  const int hy = blockIdx.y, bz = blockIdx.z;
  const int bh = bz * NHEAD + hy;
  const int q0w = blockIdx.x * 128 + wave * 16;

  // Q fragments (A-matrix 16x32, two K-chunks over HDIM=64), kept in VGPRs.
  AF qa[2];
  {
    const unsigned short* qp = qh + ((size_t)bh * SEQ + (q0w + r)) * HDIM;
#pragma unroll
    for (int kc = 0; kc < 2; kc++) {
      qa[kc].q[0] = *(const u32x4*)(qp + kc * 32 + half * 8);
      qa[kc].q[1] = *(const u32x4*)(qp + kc * 32 + 16 + half * 8);
    }
  }

  float mrow[8], lrow[8];
  CF o[4];
#pragma unroll
  for (int j = 0; j < 8; j++) { mrow[j] = -3.0e38f; lrow[j] = 0.f; }
#pragma unroll
  for (int ni = 0; ni < 4; ni++) o[ni].v = (v8f){0, 0, 0, 0, 0, 0, 0, 0};

  for (int kt0 = 0; kt0 < SEQ; kt0 += 128) {
    // K tile 128x64 (async b128), V tile transposed into [dim][key].
#pragma unroll
    for (int i = 0; i < 4; i++) {
      int idx = tid + i * 256;
      int key = idx >> 3, dw = idx & 7;
      cp16(&Kt[key][dw * 8], kh + ((size_t)bh * SEQ + kt0 + key) * HDIM + dw * 8);
    }
#pragma unroll
    for (int i = 0; i < 4; i++) {
      int idx = tid + i * 256;
      int key = idx >> 3, dw = idx & 7;
      US8 d; d.q = *(const u32x4*)(vh + ((size_t)bh * SEQ + kt0 + key) * HDIM + dw * 8);
#pragma unroll
      for (int e = 0; e < 8; e++) Vt[dw * 8 + e][key] = d.s[e];
    }
    async_wait();
    __syncthreads();

    // S = Q @ K^T for 8 key sub-tiles of 16.
    CF s[8];
#pragma unroll
    for (int kt = 0; kt < 8; kt++) {
      s[kt].v = (v8f){0, 0, 0, 0, 0, 0, 0, 0};
#pragma unroll
      for (int kc = 0; kc < 2; kc++) {
        AF bk;
        const unsigned short* p = &Kt[kt * 16 + r][kc * 32 + half * 8];
        bk.q[0] = *(const u32x4*)p;
        bk.q[1] = *(const u32x4*)(p + 16);
        s[kt].v = __builtin_amdgcn_wmma_f32_16x16x32_bf16(
            false, qa[kc].v, false, bk.v, (short)0, s[kt].v, false, false);
      }
    }

    // Online softmax: row j lives in element j (row = j + half*8), cols across
    // the 16 lanes of this half -> shfl_xor reductions with masks 1,2,4,8.
    float tmax[8];
#pragma unroll
    for (int j = 0; j < 8; j++) tmax[j] = -3.0e38f;
#pragma unroll
    for (int kt = 0; kt < 8; kt++)
#pragma unroll
      for (int j = 0; j < 8; j++) tmax[j] = fmaxf(tmax[j], s[kt].f[j]);
#pragma unroll
    for (int j = 0; j < 8; j++) tmax[j] *= 0.125f;  // 1/sqrt(64)
#pragma unroll
    for (int m = 1; m <= 8; m <<= 1)
#pragma unroll
      for (int j = 0; j < 8; j++) tmax[j] = fmaxf(tmax[j], __shfl_xor(tmax[j], m, 32));

    float cfac[8], tsum[8];
#pragma unroll
    for (int j = 0; j < 8; j++) {
      float mn = fmaxf(mrow[j], tmax[j]);
      cfac[j] = __expf(mrow[j] - mn);
      mrow[j] = mn;
      tsum[j] = 0.f;
    }
#pragma unroll
    for (int kt = 0; kt < 8; kt++)
#pragma unroll
      for (int j = 0; j < 8; j++) {
        float p = __expf(s[kt].f[j] * 0.125f - mrow[j]);
        s[kt].f[j] = p;
        tsum[j] += p;
      }
#pragma unroll
    for (int m = 1; m <= 8; m <<= 1)
#pragma unroll
      for (int j = 0; j < 8; j++) tsum[j] += __shfl_xor(tsum[j], m, 32);
#pragma unroll
    for (int j = 0; j < 8; j++) lrow[j] = lrow[j] * cfac[j] + tsum[j];
#pragma unroll
    for (int ni = 0; ni < 4; ni++)
#pragma unroll
      for (int j = 0; j < 8; j++) o[ni].f[j] *= cfac[j];

    // Stage P (bf16) into this wave's LDS scratch in [query][key] layout.
#pragma unroll
    for (int kt = 0; kt < 8; kt++)
#pragma unroll
      for (int j = 0; j < 8; j++)
        Ps[wave][j + half * 8][kt * 16 + r] = f2bf(s[kt].f[j]);
    DEV_ASM("s_wait_dscnt 0x0");  // same-wave LDS produce -> consume

    // O += P @ V : A-frags from Ps, B-frags from transposed Vt.
#pragma unroll
    for (int kc = 0; kc < 4; kc++) {
      AF pa;
      const unsigned short* pp = &Ps[wave][r][kc * 32 + half * 8];
      pa.q[0] = *(const u32x4*)pp;
      pa.q[1] = *(const u32x4*)(pp + 16);
#pragma unroll
      for (int ni = 0; ni < 4; ni++) {
        AF vb;
        const unsigned short* p = &Vt[ni * 16 + r][kc * 32 + half * 8];
        vb.q[0] = *(const u32x4*)p;
        vb.q[1] = *(const u32x4*)(p + 16);
        o[ni].v = __builtin_amdgcn_wmma_f32_16x16x32_bf16(
            false, pa.v, false, vb.v, (short)0, o[ni].v, false, false);
      }
    }
    __syncthreads();
  }

  // Normalize and store: out[b, t, h*64 + d], head-major concat as reference.
#pragma unroll
  for (int ni = 0; ni < 4; ni++)
#pragma unroll
    for (int i = 0; i < 8; i++) {
      const int t = q0w + i + half * 8;
      const float val = o[ni].f[i] / lrow[i];
      outp[((size_t)(bz * SEQ + t)) * CDIM + hy * HDIM + ni * 16 + r] = f2bf(val);
    }
}

// ---------------------------------------------------------------------------
// Host-side launch (workspace needs ~200 MB).
// ---------------------------------------------------------------------------
extern "C" void kernel_launch(void* const* d_in, const int* in_sizes, int n_in,
                              void* d_out, int out_size, void* d_ws, size_t ws_size,
                              hipStream_t stream) {
  const float* x     = (const float*)d_in[0];
  const float* ln1_g = (const float*)d_in[1];
  const float* ln1_b = (const float*)d_in[2];
  const float* ln2_g = (const float*)d_in[3];
  const float* ln2_b = (const float*)d_in[4];
  const float* wq = (const float*)d_in[5];  const float* bq = (const float*)d_in[6];
  const float* wk = (const float*)d_in[7];  const float* bk = (const float*)d_in[8];
  const float* wv = (const float*)d_in[9];  const float* bv = (const float*)d_in[10];
  const float* wo = (const float*)d_in[11]; const float* bo = (const float*)d_in[12];
  const float* w1 = (const float*)d_in[13]; const float* b1 = (const float*)d_in[14];
  const float* bn1_g = (const float*)d_in[15]; const float* bn1_b = (const float*)d_in[16];
  const float* bn1_m = (const float*)d_in[17]; const float* bn1_v = (const float*)d_in[18];
  const float* w2 = (const float*)d_in[19]; const float* b2 = (const float*)d_in[20];
  const float* bn2_g = (const float*)d_in[21]; const float* bn2_b = (const float*)d_in[22];
  const float* bn2_m = (const float*)d_in[23]; const float* bn2_v = (const float*)d_in[24];
  (void)in_sizes; (void)n_in; (void)out_size; (void)ws_size;

  char* ws = (char*)d_ws;
  auto take = [&](size_t bytes) -> char* {
    char* p = ws; ws += (bytes + 255) & ~(size_t)255; return p;
  };
  unsigned short* wq_bf = (unsigned short*)take((size_t)CDIM * CDIM * 2);
  unsigned short* wk_bf = (unsigned short*)take((size_t)CDIM * CDIM * 2);
  unsigned short* wv_bf = (unsigned short*)take((size_t)CDIM * CDIM * 2);
  unsigned short* wo_bf = (unsigned short*)take((size_t)CDIM * CDIM * 2);
  unsigned short* w1_bf = (unsigned short*)take((size_t)CDIM * HID * 2);
  unsigned short* w2_bf = (unsigned short*)take((size_t)HID * CDIM * 2);
  unsigned short* lnx   = (unsigned short*)take((size_t)N_TOK * CDIM * 2);  // reused LN1/LN2
  unsigned short* q_hd  = (unsigned short*)take((size_t)N_TOK * CDIM * 2);
  unsigned short* k_hd  = (unsigned short*)take((size_t)N_TOK * CDIM * 2);
  unsigned short* v_hd  = (unsigned short*)take((size_t)N_TOK * CDIM * 2);
  unsigned short* att   = (unsigned short*)take((size_t)N_TOK * CDIM * 2);
  float*          x1    = (float*)take((size_t)N_TOK * CDIM * 4);
  unsigned short* h_bf  = (unsigned short*)take((size_t)N_TOK * HID * 2);

  // 1) Weights to bf16.
  const int CC = CDIM * CDIM, CH = CDIM * HID;
  hipLaunchKernelGGL(k_cvt_bf16, dim3((CC + 255) / 256), dim3(256), 0, stream, wq, wq_bf, CC);
  hipLaunchKernelGGL(k_cvt_bf16, dim3((CC + 255) / 256), dim3(256), 0, stream, wk, wk_bf, CC);
  hipLaunchKernelGGL(k_cvt_bf16, dim3((CC + 255) / 256), dim3(256), 0, stream, wv, wv_bf, CC);
  hipLaunchKernelGGL(k_cvt_bf16, dim3((CC + 255) / 256), dim3(256), 0, stream, wo, wo_bf, CC);
  hipLaunchKernelGGL(k_cvt_bf16, dim3((CH + 255) / 256), dim3(256), 0, stream, w1, w1_bf, CH);
  hipLaunchKernelGGL(k_cvt_bf16, dim3((CH + 255) / 256), dim3(256), 0, stream, w2, w2_bf, CH);

  // 2) LN1.
  hipLaunchKernelGGL(k_layernorm, dim3(N_TOK), dim3(256), 0, stream, x, ln1_g, ln1_b, lnx);

  // 3) QKV projections -> head-major bf16.
  dim3 gC(CDIM / BN, N_TOK / BM);
  hipLaunchKernelGGL((k_gemm_bf16<0>), gC, dim3(256), 0, stream,
                     lnx, wq_bf, bq, nullptr, nullptr, nullptr, nullptr, nullptr,
                     (void*)q_hd, CDIM, CDIM);
  hipLaunchKernelGGL((k_gemm_bf16<0>), gC, dim3(256), 0, stream,
                     lnx, wk_bf, bk, nullptr, nullptr, nullptr, nullptr, nullptr,
                     (void*)k_hd, CDIM, CDIM);
  hipLaunchKernelGGL((k_gemm_bf16<0>), gC, dim3(256), 0, stream,
                     lnx, wv_bf, bv, nullptr, nullptr, nullptr, nullptr, nullptr,
                     (void*)v_hd, CDIM, CDIM);

  // 4) Flash attention.
  hipLaunchKernelGGL(k_flash_attn, dim3(SEQ / 128, NHEAD, BATCH), dim3(256), 0, stream,
                     q_hd, k_hd, v_hd, att);

  // 5) Output projection + residual -> x1 (f32).
  hipLaunchKernelGGL((k_gemm_bf16<1>), gC, dim3(256), 0, stream,
                     att, wo_bf, bo, nullptr, nullptr, nullptr, nullptr, x,
                     (void*)x1, CDIM, CDIM);

  // 6) LN2 on x1.
  hipLaunchKernelGGL(k_layernorm, dim3(N_TOK), dim3(256), 0, stream, x1, ln2_g, ln2_b, lnx);

  // 7) MLP up-proj: bn1 + exact GELU -> bf16.
  dim3 gH(HID / BN, N_TOK / BM);
  hipLaunchKernelGGL((k_gemm_bf16<2>), gH, dim3(256), 0, stream,
                     lnx, w1_bf, b1, bn1_g, bn1_b, bn1_m, bn1_v, nullptr,
                     (void*)h_bf, CDIM, HID);

  // 8) MLP down-proj: bn2 + residual -> d_out (f32).
  hipLaunchKernelGGL((k_gemm_bf16<3>), gC, dim3(256), 0, stream,
                     h_bf, w2_bf, b2, bn2_g, bn2_b, bn2_m, bn2_v, x1,
                     d_out, HID, CDIM);
}
